// SoftTree_6949257085098
// MI455X (gfx1250) — matrix operand order repaired and checked by reading, not compiled, not run on hardware
//
#include <hip/hip_runtime.h>
#include <hip/hip_bf16.h>

typedef __attribute__((ext_vector_type(16))) __bf16 v16bf;
typedef __attribute__((ext_vector_type(8)))  float  v8f;
typedef unsigned int u32x4 __attribute__((ext_vector_type(4)));
typedef int          i32x4 __attribute__((ext_vector_type(4)));
typedef int          i32x8 __attribute__((ext_vector_type(8)));

#define BATCH 4096
#define INF   512
#define OUTF  64
#define LEAVES 256
#define GATES 255
#define LCHUNK 64   // leaves per l-chunk (4 chunks)

// ---------------------------------------------------------------------------
// Kernel 1: gatings = sigmoid(x @ gw + gb); leaf probs p[b,l] = prod over path
// ---------------------------------------------------------------------------
__global__ __launch_bounds__(256) void gate_leaf_kernel(
    const float* __restrict__ x, const float* __restrict__ gw,
    const float* __restrict__ gb, float* __restrict__ p)
{
    __shared__ float xs[8][INF];
    __shared__ float gs[8][256];
    const int wave = threadIdx.x >> 5;
    const int lane = threadIdx.x & 31;
    const int row  = blockIdx.x * 8 + wave;

    for (int k = lane; k < INF; k += 32) xs[wave][k] = x[(size_t)row * INF + k];
    __syncthreads();

    for (int g = lane; g < GATES; g += 32) {
        float acc = gb[g];
        for (int k = 0; k < INF; ++k)
            acc = fmaf(xs[wave][k], gw[(size_t)k * GATES + g], acc);
        gs[wave][g] = 1.0f / (1.0f + __expf(-acc));
    }
    __syncthreads();

    #pragma unroll
    for (int li = 0; li < 8; ++li) {
        int l = lane * 8 + li;
        float prob = 1.0f;
        int index = 1;
        #pragma unroll
        for (int j = 0; j < 8; ++j) {
            int bit = (l >> (7 - j)) & 1;           // MSB-first, as binary_repr
            float g = gs[wave][index - 1];
            prob *= bit ? (1.0f - g) : g;
            index = 2 * index + bit;
        }
        p[(size_t)row * LEAVES + l] = prob;
    }
}

// ---------------------------------------------------------------------------
// Kernel 2: repack pw (f32, [o][k][l]) -> pwp (bf16) in WMMA-B fragment order.
// ---------------------------------------------------------------------------
__global__ __launch_bounds__(256) void pack_pw_kernel(
    const float* __restrict__ pw, unsigned short* __restrict__ pwp)
{
    const int tid  = blockIdx.x * blockDim.x + threadIdx.x; // 256*16*4*32 threads
    const int lane = tid & 31;
    const int tile = tid >> 5;              // ((l*16)+kk)*4 + ot
    const int ot = tile & 3;
    const int kk = (tile >> 2) & 15;
    const int l  = tile >> 6;
    const int o  = ot * 16 + (lane & 15);
    const int kb = (lane >> 4) * 16;

    __align__(16) unsigned short v[16];
    #pragma unroll
    for (int e = 0; e < 16; ++e) {
        int k = kk * 32 + kb + e;
        float f = pw[((size_t)o * INF + k) * LEAVES + l];
        unsigned u = __builtin_bit_cast(unsigned, f);
        v[e] = (unsigned short)(u >> 16);   // f32 -> bf16 (truncate)
    }
    uint4* dst = (uint4*)pwp + (size_t)tile * 64 + lane * 2;
    dst[0] = *(const uint4*)&v[0];
    dst[1] = *(const uint4*)&v[8];
}

// ---------------------------------------------------------------------------
// TDM helper: issue a contiguous 4 KB (1024 dword) load into LDS.
// D# per CDNA5 ISA ch8: group0 {count=1, lds_addr, global_addr, type=2},
// group1 {data_size=4B, tensor_dim0=1024, tile_dim0=1024, tile_dim1=1}.
// ---------------------------------------------------------------------------
__device__ __forceinline__ void tdm_load_4kb(unsigned lds_addr, const void* gptr)
{
    unsigned long long ga = (unsigned long long)gptr;
    u32x4 g0;
    g0[0] = 1u;                                   // count=1 valid descriptor
    g0[1] = lds_addr;                             // LDS byte address
    g0[2] = (unsigned)ga;                         // global_addr[31:0]
    g0[3] = (unsigned)(ga >> 32) | (2u << 30);    // global_addr[56:32] | type=2

    i32x8 g1;
    g1[0] = 0x00020000;        // workgroup_mask=0, data_size=2 (4 bytes)
    g1[1] = (int)(1024u << 16);// tensor_dim0[15:0]=1024 in bits[31:16]
    g1[2] = (int)(1u << 16);   // tensor_dim0 hi=0 | tensor_dim1=1
    g1[3] = (int)(1024u << 16);// tensor_dim1 hi=0 | tile_dim0=1024
    g1[4] = 1;                 // tile_dim1=1, tile_dim2=0
    g1[5] = 1024;              // tensor_dim0_stride lo
    g1[6] = (int)(1024u << 16);// stride0 hi=0 | tensor_dim1_stride lo=1024
    g1[7] = 0;

    i32x4 gz4 = {0, 0, 0, 0};
    i32x8 gz8 = {0, 0, 0, 0, 0, 0, 0, 0};
    __builtin_amdgcn_tensor_load_to_lds(g0, g1, gz4, gz4, gz8, 0);
}

// ---------------------------------------------------------------------------
// Kernel 3: main fused GEMM. Wave owns a 16-row batch tile x all 64 outputs.
// B slabs (4 KB per leaf) streamed into double-buffered LDS via the Tensor
// Data Mover; compute loop touches LDS only (ds_load_b128 for A and B).
// ---------------------------------------------------------------------------
__global__ __launch_bounds__(64) void soft_tree_main_kernel(
    const float* __restrict__ x, const float* __restrict__ p,
    const unsigned short* __restrict__ pwp, float* __restrict__ partial)
{
    __shared__ __align__(16) unsigned short xs[2][16 * INF];   // bf16 rows, 32 KB
    __shared__ __align__(16) float ps[2][LCHUNK][16];          // p tile, 8 KB
    __shared__ __align__(16) uint4 bstage[2][2][256];          // 2 waves x 2 bufs x 4 KB

    const int wave = threadIdx.x >> 5;
    const int lane = threadIdx.x & 31;
    const int btile = blockIdx.x * 2 + wave;      // 0..255
    const int rowBase = btile * 16;
    const int lBase = blockIdx.y * LCHUNK;

    // Stage x tile (16 x 512) as bf16 into LDS
    for (int t = 0; t < 256; ++t) {
        int idx = t * 32 + lane;
        int r = idx >> 9, k = idx & 511;
        float f = x[(size_t)(rowBase + r) * INF + k];
        unsigned u = __builtin_bit_cast(unsigned, f);
        xs[wave][r * INF + k] = (unsigned short)(u >> 16);
    }
    // Stage p tile (16 rows x LCHUNK leaves), row-fastest for vector reads
    for (int t = 0; t < 32; ++t) {
        int idx = t * 32 + lane;
        int r = idx >> 6, l0 = idx & 63;
        ps[wave][l0][r] = p[(size_t)(rowBase + r) * LEAVES + lBase + l0];
    }
    __syncthreads();

    const int hi = lane >> 4;
    const uint4* xrow = (const uint4*)&xs[wave][(lane & 15) * INF];

    // TDM prologue: prefetch B slab for first leaf into buffer 0
    tdm_load_4kb((unsigned)(size_t)&bstage[wave][0][0],
                 (const char*)pwp + (size_t)lBase * 4096);

    v8f acc[4] = {};

    #pragma unroll 1
    for (int l0 = 0; l0 < LCHUNK; ++l0) {
        // prefetch next leaf's B slab into the other buffer (clamped at end)
        const int lNext = (l0 + 1 < LCHUNK) ? (lBase + l0 + 1) : (lBase + l0);
        tdm_load_4kb((unsigned)(size_t)&bstage[wave][(l0 + 1) & 1][0],
                     (const char*)pwp + (size_t)lNext * 4096);
        // current buffer's TDM (issued last iteration) is complete when <=1 remains
        __builtin_amdgcn_s_wait_tensorcnt(1);

        const uint4* bq = &bstage[wave][l0 & 1][0];

        // this lane's 8 leaf probabilities (rows j + 8*hi)
        const float* pr = &ps[wave][l0][hi * 8];
        float p8[8];
        #pragma unroll
        for (int j = 0; j < 8; ++j) p8[j] = pr[j];

        v8f tmp[4] = {};
        #pragma unroll 4
        for (int kk = 0; kk < 16; ++kk) {
            // A fragment (16x32 bf16): two ds_load_b128, contiguous k = 16*hi + 0..15
            union { v16bf v; uint4 q[2]; } a;
            a.q[0] = xrow[kk * 4 + 2 * hi];       // K_local = 16*hi + 0..7
            a.q[1] = xrow[kk * 4 + 2 * hi + 1];   // K_local = 16*hi + 8..15
            #pragma unroll
            for (int ot = 0; ot < 4; ++ot) {
                union { v16bf v; uint4 q[2]; } b;
                const uint4* src = bq + ((kk * 4 + ot) * 32 + lane) * 2;
                b.q[0] = src[0];
                b.q[1] = src[1];
                tmp[ot] = __builtin_amdgcn_wmma_f32_16x16x32_bf16(
                    false, a.v, false, b.v, (short)0, tmp[ot], false, false);
            }
        }
        #pragma unroll
        for (int ot = 0; ot < 4; ++ot)
            #pragma unroll
            for (int j = 0; j < 8; ++j)
                acc[ot][j] = fmaf(tmp[ot][j], p8[j], acc[ot][j]);
    }

    // Store partial (per l-chunk). C layout: VGPR j -> M = j + 8*hi, lane&15 = N
    const size_t outBase = (size_t)blockIdx.y * BATCH * OUTF;
    const int col0 = lane & 15;
    #pragma unroll
    for (int ot = 0; ot < 4; ++ot)
        #pragma unroll
        for (int j = 0; j < 8; ++j) {
            int row = rowBase + j + 8 * hi;
            partial[outBase + (size_t)row * OUTF + ot * 16 + col0] = acc[ot][j];
        }
    // trailing TDM drains via S_ENDPGM implicit wait-idle
}

// ---------------------------------------------------------------------------
// Kernel 4: out[b,o] = sum_c partial[c][b,o] + dot(p[b,:], pb[o,:])
// ---------------------------------------------------------------------------
__global__ __launch_bounds__(256) void reduce_kernel(
    const float* __restrict__ partial, const float* __restrict__ p,
    const float* __restrict__ pb, float* __restrict__ out)
{
    const int t = blockIdx.x * blockDim.x + threadIdx.x;
    if (t >= BATCH * OUTF) return;
    const int b = t >> 6, o = t & 63;
    float s = 0.0f;
    #pragma unroll
    for (int c = 0; c < 4; ++c) s += partial[(size_t)c * BATCH * OUTF + t];
    const float* prow  = p  + (size_t)b * LEAVES;
    const float* pbrow = pb + (size_t)o * LEAVES;
    for (int l = 0; l < LEAVES; ++l) s = fmaf(prow[l], pbrow[l], s);
    out[t] = s;
}

// ---------------------------------------------------------------------------
extern "C" void kernel_launch(void* const* d_in, const int* in_sizes, int n_in,
                              void* d_out, int out_size, void* d_ws, size_t ws_size,
                              hipStream_t stream)
{
    (void)in_sizes; (void)n_in; (void)out_size; (void)ws_size;
    const float* x  = (const float*)d_in[0];
    const float* gw = (const float*)d_in[1];
    const float* gb = (const float*)d_in[2];
    const float* pw = (const float*)d_in[3];
    const float* pb = (const float*)d_in[4];
    // d_in[5] leaf_idx, d_in[6] leaf_bit: recomputed arithmetically in-kernel

    // workspace layout
    float*          p       = (float*)d_ws;                                   // 4 MB
    unsigned short* pwp     = (unsigned short*)((char*)d_ws + 4u*1024*1024);  // 16 MB
    float*          partial = (float*)((char*)d_ws + 20u*1024*1024);          // 4 MB

    gate_leaf_kernel<<<BATCH / 8, 256, 0, stream>>>(x, gw, gb, p);
    pack_pw_kernel<<<(LEAVES * 16 * 4 * 32) / 256, 256, 0, stream>>>(pw, pwp);
    soft_tree_main_kernel<<<dim3(128, 4), 64, 0, stream>>>(x, p, pwp, partial);
    reduce_kernel<<<(BATCH * OUTF) / 256, 256, 0, stream>>>(partial, p, pb, (float*)d_out);
}
